// TripletLoss_40114994544726
// MI455X (gfx1250) — compile-verified, matrix operand order checked by reading, not run on hardware
//
#include <hip/hip_runtime.h>
#include <hip/hip_bf16.h>

// ---------------------------------------------------------------------------
// Triplet loss (batch-hard mining, cosine distance) for MI455X / gfx1250.
//   X: 8192 x 256 fp32, labels: 8192 int32  ->  scalar fp32 loss
// Strategy:
//   1) normalize rows -> f16 matrix in workspace (fits in 192MB L2: 4MB)
//   2) fused X * X^T via v_wmma_f32_16x16x32_f16, per-tile hard mining
//   3) global float atomics for (loss_sum, valid_count), finalize kernel
// Workspace layout: [0..7] two float accumulators, [256 ...] f16 matrix (4MB).
// ---------------------------------------------------------------------------

typedef __attribute__((ext_vector_type(16))) _Float16 v16h;
typedef __attribute__((ext_vector_type(8)))  _Float16 v8h;
typedef __attribute__((ext_vector_type(8)))  float    v8f;
typedef __attribute__((ext_vector_type(4)))  float    v4f;

#define NROW 8192
#define NDIM 256
#define BIGF 1.0e9f

// --------------------------- kernel 0: zero accumulators -------------------
__global__ void tl_zero_kernel(float* acc) {
    if (threadIdx.x < 2) acc[threadIdx.x] = 0.0f;
}

// --------------------------- kernel 1: row-normalize -> f16 ----------------
// 256 threads = 8 waves; one wave per row; each lane owns 8 contiguous cols.
__global__ __launch_bounds__(256) void tl_normalize_kernel(const float* __restrict__ x,
                                                           _Float16* __restrict__ out) {
    const int wave = threadIdx.x >> 5;
    const int lane = threadIdx.x & 31;
    const int row  = blockIdx.x * 8 + wave;

    const float* rp = x + (size_t)row * NDIM + lane * 8;
    v4f a = *(const v4f*)rp;
    v4f b = *(const v4f*)(rp + 4);

    float ss = a.x*a.x + a.y*a.y + a.z*a.z + a.w*a.w
             + b.x*b.x + b.y*b.y + b.z*b.z + b.w*b.w;
    // wave32 all-lane reduction
    #pragma unroll
    for (int m = 16; m >= 1; m >>= 1) ss += __shfl_xor(ss, m, 32);

    const float scale = 1.0f / fmaxf(sqrtf(ss), 1e-12f);

    v8h h;
    h[0] = (_Float16)(a.x * scale); h[1] = (_Float16)(a.y * scale);
    h[2] = (_Float16)(a.z * scale); h[3] = (_Float16)(a.w * scale);
    h[4] = (_Float16)(b.x * scale); h[5] = (_Float16)(b.y * scale);
    h[6] = (_Float16)(b.z * scale); h[7] = (_Float16)(b.w * scale);
    *(v8h*)(out + (size_t)row * NDIM + lane * 8) = h;
}

// --------------------------- kernel 2: fused GEMM + hard mining ------------
// grid 64 x 256 threads (8 waves). Wave w of block b owns anchor rows
// [ (b*8+w)*16 , +16 ). A-fragments (16x256 f16 = 8 x v16h) stay in VGPRs;
// sweep 512 column tiles, 8 WMMAs each, mine pos-max / neg-min in epilogue.
//
// ISA fragment layout (16-bit A, 16x32): lane L holds row M=L&15; halves
// 0..7 are K = 8*(L>>4) .. +7, halves 8..15 are K = 8*(L>>4)+16 .. +23.
// B (32x16, K x N) for X*X^T loads identically from row N of X (row-major).
// C/D f32 16x16: VGPR r, lane L -> M = r + 8*(L>>4), N = L&15.
__global__ __launch_bounds__(256, 2) void tl_mine_kernel(const _Float16* __restrict__ Xh,
                                                         const int* __restrict__ labels,
                                                         float* __restrict__ acc) {
    __shared__ int s_lab[NROW];   // 32 KB of 320 KB LDS
    for (int i = threadIdx.x; i < NROW; i += 256) s_lab[i] = labels[i];
    __syncthreads();

    const int wave = threadIdx.x >> 5;
    const int lane = threadIdx.x & 31;
    const int half = lane >> 4;       // which 16-lane half
    const int l15  = lane & 15;
    const int m0   = (blockIdx.x * 8 + wave) * 16;   // anchor strip base
    const int kb   = half * 8;        // K sub-offset inside each 32-chunk

    // Resident A fragments for rows m0..m0+15, K = 0..255
    const int arow = m0 + l15;
    v16h afrag[8];
    #pragma unroll
    for (int kk = 0; kk < 8; ++kk) {
        const _Float16* p = Xh + (size_t)arow * NDIM + kk * 32 + kb;
        v8h lo = *(const v8h*)p;
        v8h hi = *(const v8h*)(p + 16);
        #pragma unroll
        for (int i = 0; i < 8; ++i) { afrag[kk][i] = lo[i]; afrag[kk][i + 8] = hi[i]; }
    }

    // Anchor labels for the 8 M-rows this lane's C registers map to
    int labm[8];
    #pragma unroll
    for (int r = 0; r < 8; ++r) labm[r] = s_lab[m0 + r + 8 * half];

    float posv[8], negv[8];
    #pragma unroll
    for (int r = 0; r < 8; ++r) { posv[r] = -BIGF; negv[r] = BIGF; }

    // Sweep all 512 column tiles
    #pragma unroll 1
    for (int n0 = 0; n0 < NROW; n0 += 16) {
        const int brow = n0 + l15;

        v16h bfrag[8];
        #pragma unroll
        for (int kk = 0; kk < 8; ++kk) {
            const _Float16* p = Xh + (size_t)brow * NDIM + kk * 32 + kb;
            v8h lo = *(const v8h*)p;
            v8h hi = *(const v8h*)(p + 16);
            #pragma unroll
            for (int i = 0; i < 8; ++i) { bfrag[kk][i] = lo[i]; bfrag[kk][i + 8] = hi[i]; }
        }

        v8f c = {};
        #pragma unroll
        for (int kk = 0; kk < 8; ++kk) {
            c = __builtin_amdgcn_wmma_f32_16x16x32_f16(
                    /*neg_a=*/false, afrag[kk],
                    /*neg_b=*/false, bfrag[kk],
                    /*c_mod=*/(short)0, c,
                    /*reuse_a=*/false, /*reuse_b=*/false);
        }

        const int labn = s_lab[brow];
        #pragma unroll
        for (int r = 0; r < 8; ++r) {
            float d = fmaxf(1.0f - c[r], 0.0f);       // clipped cosine distance
            const int m    = m0 + r + 8 * half;
            const bool same = (labm[r] == labn);
            const bool diag = (m == brow);
            posv[r] = (same && !diag) ? fmaxf(posv[r], d) : posv[r];
            negv[r] = (!same)         ? fminf(negv[r], d) : negv[r];
        }
    }

    // Reduce across the 16 lanes sharing each M row (N is striped on lane&15)
    #pragma unroll
    for (int r = 0; r < 8; ++r) {
        #pragma unroll
        for (int m = 8; m >= 1; m >>= 1) {
            posv[r] = fmaxf(posv[r], __shfl_xor(posv[r], m, 32));
            negv[r] = fminf(negv[r], __shfl_xor(negv[r], m, 32));
        }
    }

    if (l15 == 0) {   // lanes 0 and 16 each finalize 8 anchor rows
        float lsum = 0.0f, lcnt = 0.0f;
        #pragma unroll
        for (int r = 0; r < 8; ++r) {
            const bool valid = (posv[r] > -0.5f * BIGF) && (negv[r] < 0.5f * BIGF);
            if (valid) {
                lsum += fmaxf(posv[r] - negv[r] + 1.0f, 0.0f);  // MARGIN = 1.0
                lcnt += 1.0f;
            }
        }
        atomicAdd(&acc[0], lsum);
        atomicAdd(&acc[1], lcnt);
    }
}

// --------------------------- kernel 3: finalize ----------------------------
__global__ void tl_finalize_kernel(const float* __restrict__ acc, float* __restrict__ out) {
    const float s = acc[0], c = acc[1];
    out[0] = (c > 0.0f) ? (s / fmaxf(c, 1.0f)) : 0.0f;
}

// ---------------------------------------------------------------------------
extern "C" void kernel_launch(void* const* d_in, const int* in_sizes, int n_in,
                              void* d_out, int out_size, void* d_ws, size_t ws_size,
                              hipStream_t stream) {
    const float* x      = (const float*)d_in[0];
    const int*   labels = (const int*)d_in[1];
    float*       out    = (float*)d_out;

    float*     acc = (float*)d_ws;                         // 2 floats
    _Float16*  Xh  = (_Float16*)((char*)d_ws + 256);       // 8192*256*2 = 4 MB

    tl_zero_kernel<<<1, 32, 0, stream>>>(acc);
    tl_normalize_kernel<<<NROW / 8, 256, 0, stream>>>(x, Xh);
    tl_mine_kernel<<<(NROW / 16) / 8, 256, 0, stream>>>(Xh, labels, acc);
    tl_finalize_kernel<<<1, 1, 0, stream>>>(acc, out);
}
